// DMPNNEncoderLayer_52209622450218
// MI455X (gfx1250) — compile-verified, matrix operand order checked by reading, not run on hardware
//
#include <hip/hip_runtime.h>
#include <stdint.h>

typedef __attribute__((ext_vector_type(16))) _Float16 v16h;
typedef __attribute__((ext_vector_type(8)))  _Float16 v8h;
typedef __attribute__((ext_vector_type(8)))  float    v8f;

namespace {
constexpr int kAtoms = 65536;
constexpr int kBonds = 131072;
constexpr int kMols  = 2048;
constexpr int kApm   = 32;     // atoms per molecule
constexpr int kAtomF = 133;
constexpr int kIniF  = 147;    // ATOM_F + BOND_F
constexpr int kDh    = 300;
constexpr int kDhp   = 304;    // padded hidden stride (19*16); pad cols kept == 0
constexpr int kGlob  = 128;
constexpr int kOutF  = 428;    // kDh + kGlob
constexpr int kNp    = 320;    // padded N for weights (multiple of 64)
constexpr int kKpWi  = 160;    // 147 -> pad to 5*32
constexpr int kKpWh  = 320;    // 300 -> pad to 10*32
constexpr int kKpWo  = 448;    // 433 -> pad to 14*32
constexpr int kCatF  = 433;    // 133 + 300
constexpr int kLdsK  = 40;     // LDS k-stride in halves (80B: 16B aligned, bank-spread)
}

// ---------- packing kernels (run once per launch, trivial cost) ----------

// f32 weight W[K x N] row-major -> zero-padded N-major f16 Wt[kNp x KP].
__global__ __launch_bounds__(256) void pack_weight(const float* __restrict__ W,
                                                   _Float16* __restrict__ Wt,
                                                   int K, int N, int KP) {
  int idx = blockIdx.x * 256 + threadIdx.x;
  if (idx >= kNp * KP) return;
  int n = idx / KP;
  int k = idx - n * KP;
  float v = (k < K && n < N) ? W[(size_t)k * N + n] : 0.0f;
  Wt[idx] = (_Float16)v;
}

// f_ini [kBonds x 147] f32 -> [kBonds x 160] f16, K zero-padded.
__global__ __launch_bounds__(160) void pack_f_ini(const float* __restrict__ src,
                                                  _Float16* __restrict__ dst) {
  const int r = blockIdx.x;
  const int c = threadIdx.x;                  // 0..159
  float v = (c < kIniF) ? src[(size_t)r * kIniF + c] : 0.0f;
  dst[(size_t)r * kKpWi + c] = (_Float16)v;
}

// ---------- gather kernels ----------

// m1[row] = sum_j relu(inp[mapping[row][j]])   (f32 -> f32, stride kDhp)
__global__ __launch_bounds__(256) void gather_relu_sum4(const float* __restrict__ src,
                                                        float* __restrict__ dst,
                                                        const int* __restrict__ idx) {
  const int row = blockIdx.x;
  const int i0 = idx[row * 4 + 0], i1 = idx[row * 4 + 1];
  const int i2 = idx[row * 4 + 2], i3 = idx[row * 4 + 3];
  for (int c = threadIdx.x; c < kDhp; c += 256) {
    float a = fmaxf(src[(size_t)i0 * kDhp + c], 0.0f);
    float b = fmaxf(src[(size_t)i1 * kDhp + c], 0.0f);
    float d = fmaxf(src[(size_t)i2 * kDhp + c], 0.0f);
    float e = fmaxf(src[(size_t)i3 * kDhp + c], 0.0f);
    dst[(size_t)row * kDhp + c] = (a + b) + (d + e);
  }
}

// m2[row] = sum_j m1[mapping[row][j]]  written directly as K-padded f16
// [kBonds x 320] -- this is GEMM2's A operand, no further f32 copy needed.
__global__ __launch_bounds__(256) void gather_sum4_f16(const float* __restrict__ src,
                                                       _Float16* __restrict__ dst,
                                                       const int* __restrict__ idx) {
  const int row = blockIdx.x;
  const int i0 = idx[row * 4 + 0], i1 = idx[row * 4 + 1];
  const int i2 = idx[row * 4 + 2], i3 = idx[row * 4 + 3];
  for (int c = threadIdx.x; c < kKpWh; c += 256) {
    float s = 0.0f;
    if (c < kDhp) {
      s = (src[(size_t)i0 * kDhp + c] + src[(size_t)i1 * kDhp + c]) +
          (src[(size_t)i2 * kDhp + c] + src[(size_t)i3 * kDhp + c]);
    }
    dst[(size_t)row * kKpWh + c] = (_Float16)s;
  }
}

// Fused: concat(atom_features[a], sum_j h[a2ib[a][j]]) -> f16 [kAtoms x 448],
// K zero-padded. This is GEMM3's A operand.
__global__ __launch_bounds__(256) void atom_concat_f16(const float* __restrict__ atomf,
                                                       const float* __restrict__ h,
                                                       const int* __restrict__ idx,
                                                       _Float16* __restrict__ dst) {
  const int a = blockIdx.x;
  const int i0 = idx[a * 4 + 0], i1 = idx[a * 4 + 1];
  const int i2 = idx[a * 4 + 2], i3 = idx[a * 4 + 3];
  for (int c = threadIdx.x; c < kKpWo; c += 256) {
    float v;
    if (c < kAtomF) {
      v = atomf[(size_t)a * kAtomF + c];
    } else if (c < kCatF) {
      const int cc = c - kAtomF;
      v = (h[(size_t)i0 * kDhp + cc] + h[(size_t)i1 * kDhp + cc]) +
          (h[(size_t)i2 * kDhp + cc] + h[(size_t)i3 * kDhp + cc]);
    } else {
      v = 0.0f;
    }
    dst[(size_t)a * kKpWo + c] = (_Float16)v;
  }
}

// ---------- readout ----------

__global__ __launch_bounds__(256) void readout_mean_concat(const float* __restrict__ atoms_h,
                                                           const float* __restrict__ glob,
                                                           float* __restrict__ out) {
  const int m = blockIdx.x;
  for (int c = threadIdx.x; c < kDh; c += 256) {
    float s = 0.0f;
#pragma unroll 4
    for (int i = 0; i < kApm; ++i)
      s += atoms_h[(size_t)(m * kApm + i) * kDhp + c];
    out[(size_t)m * kOutF + c] = s * (1.0f / (float)kApm);
  }
  if (threadIdx.x < kGlob)
    out[(size_t)m * kOutF + kDh + threadIdx.x] = glob[(size_t)m * kGlob + threadIdx.x];
}

// ---------- WMMA GEMM ----------

// Load one 16x32 f16 operand fragment from an LDS tile stored [row][k]:
//  lane<16 : row=lane,    k-halves {0..7} and {16..23}
//  lane>=16: row=lane-16, k-halves {8..15} and {24..31}
__device__ __forceinline__ v16h load_frag(const _Float16* tile, int lane) {
  const _Float16* p = tile + (lane & 15) * kLdsK + ((lane >> 4) << 3);
  v8h lo = *(const v8h*)p;
  v8h hi = *(const v8h*)(p + 16);
  v16h r;
#pragma unroll
  for (int i = 0; i < 8; ++i) { r[i] = lo[i]; r[i + 8] = hi[i]; }
  return r;
}

// C = A[M x KP](f16, packed) @ Wt[kNp x KP](f16, N-major).
// Block tile 128x64, 8 waves: wave w owns rows w*16..w*16+15 and all 64 cols:
// per k-step 1 A-fragment + 4 B-fragments feed 4 WMMAs (4:5 wmma:ds ratio).
// Software-pipelined global->LDS staging; MODE epilogue:
//   0 = store raw; 1 = relu(acc + addC); 2 = relu(acc + bias).
template <int MODE>
__global__ __launch_bounds__(256) void gemm_wmma(
    int KP, const _Float16* __restrict__ A, const _Float16* __restrict__ Wt,
    const float* __restrict__ addC, const float* __restrict__ bias,
    float* __restrict__ Cout) {
  __shared__ _Float16 Ald[128 * kLdsK];
  __shared__ _Float16 Bld[64 * kLdsK];

  const int tid  = threadIdx.x;
  const int lane = tid & 31;
  const int wm   = tid >> 5;        // 0..7 -> 16-row strip
  const int m0   = blockIdx.x * 128;
  const int n0   = blockIdx.y * 64;

  // A staging: 128 rows x 32 k = 4096 halves; thread covers 16 halves.
  const int trA = tid >> 1;               // 0..127
  const int tkA = (tid & 1) * 16;         // 0,16
  // B staging: 64 rows x 32 k = 2048 halves; thread covers 8 halves.
  const int trB = tid >> 2;               // 0..63
  const int tkB = (tid & 3) * 8;          // 0,8,16,24

  const _Float16* ap = A  + (size_t)(m0 + trA) * KP + tkA;  // guard-free, 16B aligned
  const _Float16* bp = Wt + (size_t)(n0 + trB) * KP + tkB;

  // Output coordinates (C/D layout: VGPR v, lane l -> M = v + (l>=16?8:0), N = l&15).
  const int rb  = m0 + wm * 16 + ((lane >> 4) << 3);
  const int cb0 = n0 + (lane & 15);

  if (MODE == 1) {
    // Pull the f32 C-addend tile toward the caches while the GEMM runs.
#pragma unroll
    for (int v = 0; v < 8; ++v)
      __builtin_prefetch(&addC[(size_t)(rb + v) * kDhp + cb0], 0, 3);
  }

  v8f c0 = {}, c1 = {}, c2 = {}, c3 = {};
  v8h ar0 = *(const v8h*)ap;
  v8h ar1 = *(const v8h*)(ap + 8);
  v8h br  = *(const v8h*)bp;

  for (int k0 = 0; k0 < KP; k0 += 32) {
    *(v8h*)&Ald[trA * kLdsK + tkA]     = ar0;
    *(v8h*)&Ald[trA * kLdsK + tkA + 8] = ar1;
    *(v8h*)&Bld[trB * kLdsK + tkB]     = br;
    __syncthreads();
    if (k0 + 32 < KP) {               // prefetch next tiles during WMMA
      ar0 = *(const v8h*)(ap + k0 + 32);
      ar1 = *(const v8h*)(ap + k0 + 40);
      br  = *(const v8h*)(bp + k0 + 32);
    }
    v16h a  = load_frag(Ald + (wm * 16) * kLdsK, lane);
    v16h b0 = load_frag(Bld,               lane);
    v16h b1 = load_frag(Bld + 16 * kLdsK,  lane);
    v16h b2 = load_frag(Bld + 32 * kLdsK,  lane);
    v16h b3 = load_frag(Bld + 48 * kLdsK,  lane);
    c0 = __builtin_amdgcn_wmma_f32_16x16x32_f16(false, a, false, b0, (short)0, c0,
                                                false, false);
    c1 = __builtin_amdgcn_wmma_f32_16x16x32_f16(false, a, false, b1, (short)0, c1,
                                                false, false);
    c2 = __builtin_amdgcn_wmma_f32_16x16x32_f16(false, a, false, b2, (short)0, c2,
                                                false, false);
    c3 = __builtin_amdgcn_wmma_f32_16x16x32_f16(false, a, false, b3, (short)0, c3,
                                                false, false);
    __syncthreads();
  }

#pragma unroll
  for (int v = 0; v < 8; ++v) {
    const int gr = rb + v;
#pragma unroll
    for (int s = 0; s < 4; ++s) {
      const int gc = cb0 + s * 16;
      if (gc >= kDhp) continue;       // only N-blocks at 256..319 clip
      float val = (s == 0) ? c0[v] : (s == 1) ? c1[v] : (s == 2) ? c2[v] : c3[v];
      if (MODE == 1) val = fmaxf(val + addC[(size_t)gr * kDhp + gc], 0.0f);
      if (MODE == 2) val = fmaxf(val + ((gc < kDh) ? bias[gc] : 0.0f), 0.0f);
      Cout[(size_t)gr * kDhp + gc] = val;
    }
  }
}

// ---------- launch ----------

extern "C" void kernel_launch(void* const* d_in, const int* in_sizes, int n_in,
                              void* d_out, int out_size, void* d_ws, size_t ws_size,
                              hipStream_t stream) {
  (void)in_sizes; (void)n_in; (void)out_size; (void)ws_size;

  const float* atom_features = (const float*)d_in[0];   // [65536 x 133]
  const float* f_ini         = (const float*)d_in[1];   // [131072 x 147]
  const float* globf         = (const float*)d_in[2];   // [2048 x 128]
  const float* W_i           = (const float*)d_in[3];   // [147 x 300]
  const float* W_h           = (const float*)d_in[4];   // [300 x 300]
  const float* W_o           = (const float*)d_in[5];   // [433 x 300]
  const float* b_o           = (const float*)d_in[6];   // [300]
  const int*   a2ib          = (const int*)d_in[7];     // [65536 x 4]
  const int*   mapping       = (const int*)d_in[8];     // [131072 x 4]
  float* out = (float*)d_out;                           // [2048 x 428]

  // Workspace (~403 MB): two f32 bond buffers + one shared f16 A-staging buffer.
  const size_t SZ   = (size_t)kBonds * kDhp;            // floats per f32 buffer
  const size_t SZ16 = (size_t)kBonds * kKpWh;           // halves: max A16 size (84MB)
  float*    buf_inp = (float*)d_ws;                     // inp; later atoms_h
  float*    buf_m1  = buf_inp + SZ;                     // m1; later h
  _Float16* bufA16  = (_Float16*)(buf_m1 + SZ);         // A-operand staging (f16)
  _Float16* WiT     = bufA16 + SZ16;
  _Float16* WhT     = WiT + (size_t)kNp * kKpWi;
  _Float16* WoT     = WhT + (size_t)kNp * kKpWh;

  pack_weight<<<(kNp * kKpWi + 255) / 256, 256, 0, stream>>>(W_i, WiT, kIniF, kDh, kKpWi);
  pack_weight<<<(kNp * kKpWh + 255) / 256, 256, 0, stream>>>(W_h, WhT, kDh, kDh, kKpWh);
  pack_weight<<<(kNp * kKpWo + 255) / 256, 256, 0, stream>>>(W_o, WoT, kCatF, kDh, kKpWo);

  dim3 gBond(kBonds / 128, kNp / 64);   // (1024, 5)
  dim3 gAtom(kAtoms / 128, kNp / 64);   // (512, 5)

  // A1 = pad16(f_ini); inp = A1 @ W_i   (pre-relu; pad cols exactly 0)
  pack_f_ini<<<kBonds, 160, 0, stream>>>(f_ini, bufA16);
  gemm_wmma<0><<<gBond, 256, 0, stream>>>(kKpWi, bufA16, WiT, nullptr, nullptr, buf_inp);
  // m1 = gather4(relu(inp)); A2 = pad16(gather4(m1))  [first W_h GEMM is dead code]
  gather_relu_sum4<<<kBonds, 256, 0, stream>>>(buf_inp, buf_m1, mapping);
  gather_sum4_f16<<<kBonds, 256, 0, stream>>>(buf_m1, bufA16, mapping);
  // h = relu(inp + A2 @ W_h) -> overwrite buf_m1
  gemm_wmma<1><<<gBond, 256, 0, stream>>>(kKpWh, bufA16, WhT, buf_inp, nullptr, buf_m1);
  // A3 = pad16(concat(atom_features, gather4(h)))
  atom_concat_f16<<<kAtoms, 256, 0, stream>>>(atom_features, buf_m1, a2ib, bufA16);
  // atoms_h = relu(A3 @ W_o + b_o) -> overwrite buf_inp
  gemm_wmma<2><<<gAtom, 256, 0, stream>>>(kKpWo, bufA16, WoT, nullptr, b_o, buf_inp);
  // per-molecule mean + concat globals
  readout_mean_concat<<<kMols, 256, 0, stream>>>(buf_inp, globf, out);
}
